// CarFormerAttention_89515708383857
// MI455X (gfx1250) — matrix operand order repaired and checked by reading, not compiled, not run on hardware
//
#include <hip/hip_runtime.h>

// Problem dims
#define Bn   2
#define Sn   2048
#define Hn   16
#define Dn   64
#define HIDn 1024
#define Mn   (Bn * Sn)   // 4096

typedef __bf16 bf16;
typedef __attribute__((ext_vector_type(8)))  bf16  v8bf;
typedef __attribute__((ext_vector_type(16))) bf16  v16bf;
typedef __attribute__((ext_vector_type(8)))  float v8f;

static __device__ __forceinline__ v16bf cat16(v8bf lo, v8bf hi) {
  return __builtin_shufflevector(lo, hi, 0,1,2,3,4,5,6,7,8,9,10,11,12,13,14,15);
}

// A-matrix fragment (16-bit 16x32): lane holds row (lane&15); two contiguous
// 8-element chunks at K = koff+8*half and K = koff+16+8*half (ISA layout table).
static __device__ __forceinline__ v16bf fragA(const bf16* __restrict__ row, int koff, int half) {
  v8bf lo = *(const v8bf*)(row + koff + 8 * half);
  v8bf hi = *(const v8bf*)(row + koff + 16 + 8 * half);
  return cat16(lo, hi);
}

// B-matrix fragment (16-bit 32x16): lane holds column (lane&15); two contiguous
// 8-element chunks at K = koff+8*half and K = koff+16+8*half. `row` points to opT[n][k].
static __device__ __forceinline__ v16bf fragB(const bf16* __restrict__ row, int koff, int half) {
  v8bf lo = *(const v8bf*)(row + koff + 8 * half);
  v8bf hi = *(const v8bf*)(row + koff + 16 + 8 * half);
  return cat16(lo, hi);
}

static __device__ __forceinline__ v8f wmma_bf16(v16bf a, v16bf b, v8f c) {
  return __builtin_amdgcn_wmma_f32_16x16x32_bf16(false, a, false, b, (short)0, c, false, false);
}

// ---------------------------------------------------------------- K1: converts
__global__ void k_cvt(const float* __restrict__ in, bf16* __restrict__ out, int n) {
  int i = blockIdx.x * blockDim.x + threadIdx.x;
  int stride = gridDim.x * blockDim.x;
  for (; i < n; i += stride) out[i] = (bf16)in[i];
}

// wt[n*1024 + k] = (bf16) w[k*1024 + n]
__global__ void k_wT(const float* __restrict__ w, bf16* __restrict__ wt) {
  int i = blockIdx.x * blockDim.x + threadIdx.x;   // exactly 1024*1024 threads
  int n = i >> 10;
  int k = i & 1023;
  wt[i] = (bf16)w[(size_t)k * HIDn + n];
}

// ------------------------------------------- K2: QKV GEMM + info-add + RoPE + layout
// grid(16, 16, 3), block 256 (8 waves). Wave: 32 seq rows x one full head (64 cols)
// -> B-fragments reused across two A-row tiles (8 WMMAs per 12 b128 loads).
__global__ void k_qkv(const bf16* __restrict__ X,
                      const bf16* __restrict__ WqT, const bf16* __restrict__ WkT,
                      const bf16* __restrict__ WvT, const float* __restrict__ info,
                      bf16* __restrict__ Qb, bf16* __restrict__ Kb, bf16* __restrict__ Vt) {
  const int lane = threadIdx.x & 31;
  const int wv   = threadIdx.x >> 5;
  const int half = lane >> 4;
  const int nl   = lane & 15;
  const int mt   = blockIdx.x * 8 + wv;     // 0..127 (32-row tiles)
  const int h    = blockIdx.y;
  const int which = blockIdx.z;             // 0=q 1=k 2=v
  const bf16* WT = (which == 0) ? WqT : ((which == 1) ? WkT : WvT);

  const int r0 = mt * 32;
  const bf16* arow0 = X + (size_t)(r0 + nl) * HIDn;
  const bf16* arow1 = X + (size_t)(r0 + 16 + nl) * HIDn;
  const bf16* brow[4];
  #pragma unroll
  for (int t = 0; t < 4; t++) brow[t] = WT + (size_t)(h * 64 + t * 16 + nl) * HIDn;

  v8f acc[8] = {};   // [rt*4 + t]
  for (int ko = 0; ko < HIDn; ko += 32) {
    v16bf a0 = fragA(arow0, ko, half);
    v16bf a1 = fragA(arow1, ko, half);
    #pragma unroll
    for (int t = 0; t < 4; t++) {
      v16bf b = fragB(brow[t], ko, half);
      acc[t]     = wmma_bf16(a0, b, acc[t]);
      acc[4 + t] = wmma_bf16(a1, b, acc[4 + t]);
    }
  }

  const int bb = r0 >> 11;          // batch
  const int s0 = r0 & (Sn - 1);     // seq offset of tile
  const int bh = bb * Hn + h;

  if (which == 2) {
    // V: store transposed (B,H,D,S) so PV B-fragments are contiguous.
    #pragma unroll
    for (int t = 0; t < 4; t++) {
      bf16* d = Vt + ((size_t)bh * 64 + (t * 16 + nl)) * Sn;
      #pragma unroll
      for (int rt = 0; rt < 2; rt++) {
        #pragma unroll
        for (int g = 0; g < 8; g++) {
          int srow = s0 + rt * 16 + g + 8 * half;
          d[srow] = (bf16)acc[rt * 4 + t][g];
        }
      }
    }
  } else {
    bf16* dstQK = ((which == 0) ? Qb : Kb) + (size_t)bh * Sn * 64;
    // inv_freq[j] = theta^(-2j/64); cos/sin identical for d and d+32.
    const float kln = 9.2103403719761836f / 64.0f;   // ln(10000)/64
    float inv0 = __expf(-(float)(2 * nl) * kln);          // j = nl       (tiles 0,2)
    float inv1 = __expf(-(float)(2 * (nl + 16)) * kln);   // j = nl + 16  (tiles 1,3)
    #pragma unroll
    for (int rt = 0; rt < 2; rt++) {
      const float* inf = info + (size_t)(r0 + rt * 16) * HIDn + h * 64;
      #pragma unroll
      for (int g = 0; g < 8; g++) {
        int m = g + 8 * half;
        int srow = s0 + rt * 16 + m;
        float x0 = acc[rt * 4 + 0][g] + inf[(size_t)m * HIDn +  0 + nl];
        float x1 = acc[rt * 4 + 1][g] + inf[(size_t)m * HIDn + 16 + nl];
        float x2 = acc[rt * 4 + 2][g] + inf[(size_t)m * HIDn + 32 + nl];
        float x3 = acc[rt * 4 + 3][g] + inf[(size_t)m * HIDn + 48 + nl];
        float a0 = (float)srow * inv0;
        float a1 = (float)srow * inv1;
        float c0 = __cosf(a0), sn0 = __sinf(a0);
        float c1 = __cosf(a1), sn1 = __sinf(a1);
        bf16* dr = dstQK + (size_t)srow * 64;
        dr[ 0 + nl] = (bf16)(x0 * c0 - x2 * sn0);
        dr[16 + nl] = (bf16)(x1 * c1 - x3 * sn1);
        dr[32 + nl] = (bf16)(x2 * c0 + x0 * sn0);
        dr[48 + nl] = (bf16)(x3 * c1 + x1 * sn1);
      }
    }
  }
}

// ------------------------------------------- K3: causal flash attention (per b,h)
// grid(8, 32), block 256. Wave: 32 query rows; K/V fragments amortized over two
// 16-row query tiles (16 WMMAs per 16 b128 loads per 32-key step).
__global__ void k_attn(const bf16* __restrict__ Qb, const bf16* __restrict__ Kb,
                       const bf16* __restrict__ Vt, bf16* __restrict__ Ab) {
  __shared__ bf16 pbuf[8][2][16][32];   // 16 KB
  const int lane = threadIdx.x & 31;
  const int wv   = threadIdx.x >> 5;
  const int half = lane >> 4;
  const int nl   = lane & 15;
  const int it   = blockIdx.x * 8 + wv;   // 0..63 (32-row query blocks)
  const int bh   = blockIdx.y;            // 0..31
  const int i0   = it * 32;

  v16bf aQ0[2], aQ1[2];
  #pragma unroll
  for (int qt = 0; qt < 2; qt++) {
    const bf16* qrow = Qb + ((size_t)bh * Sn + i0 + qt * 16 + nl) * 64;
    aQ0[qt] = fragA(qrow,  0, half);
    aQ1[qt] = fragA(qrow, 32, half);
  }

  v8f o[8] = {};               // [qt*4 + t]
  float mrow[16], lrow[16];    // [qt*8 + g]
  #pragma unroll
  for (int g = 0; g < 16; g++) { mrow[g] = -1e30f; lrow[g] = 0.0f; }

  const int jmax = i0 + 32;       // exclusive causal bound for this 32-row block
  const float scale = 0.125f;     // 1/sqrt(64)

  for (int j0 = 0; j0 < jmax; j0 += 32) {
    // shared K fragments for both query tiles
    const bf16* krow0 = Kb + ((size_t)bh * Sn + j0 + nl) * 64;
    const bf16* krow1 = Kb + ((size_t)bh * Sn + j0 + 16 + nl) * 64;
    v16bf bK00 = fragB(krow0,  0, half);
    v16bf bK01 = fragB(krow0, 32, half);
    v16bf bK10 = fragB(krow1,  0, half);
    v16bf bK11 = fragB(krow1, 32, half);

    #pragma unroll
    for (int qt = 0; qt < 2; qt++) {
      v8f sc0 = {}, sc1 = {};
      sc0 = wmma_bf16(aQ0[qt], bK00, sc0);
      sc0 = wmma_bf16(aQ1[qt], bK01, sc0);
      sc1 = wmma_bf16(aQ0[qt], bK10, sc1);
      sc1 = wmma_bf16(aQ1[qt], bK11, sc1);
      // online softmax over this 32-key block
      #pragma unroll
      for (int g = 0; g < 8; g++) {
        int row = i0 + qt * 16 + g + 8 * half;
        float s0v = sc0[g] * scale;
        if (j0 + nl > row) s0v = -1e30f;
        float s1v = sc1[g] * scale;
        if (j0 + 16 + nl > row) s1v = -1e30f;
        float mx = fmaxf(s0v, s1v);
        #pragma unroll
        for (int off = 1; off < 16; off <<= 1) mx = fmaxf(mx, __shfl_xor(mx, off, 32));
        float mnew  = fmaxf(mrow[qt * 8 + g], mx);
        float alpha = __expf(mrow[qt * 8 + g] - mnew);
        mrow[qt * 8 + g] = mnew;
        float e0 = __expf(s0v - mnew);
        float e1 = __expf(s1v - mnew);
        float rs = e0 + e1;
        #pragma unroll
        for (int off = 1; off < 16; off <<= 1) rs += __shfl_xor(rs, off, 32);
        lrow[qt * 8 + g] = lrow[qt * 8 + g] * alpha + rs;
        o[qt * 4 + 0][g] *= alpha;
        o[qt * 4 + 1][g] *= alpha;
        o[qt * 4 + 2][g] *= alpha;
        o[qt * 4 + 3][g] *= alpha;
        // C-layout -> row-major P tile in LDS (bf16)
        pbuf[wv][qt][g + 8 * half][ 0 + nl] = (bf16)e0;
        pbuf[wv][qt][g + 8 * half][16 + nl] = (bf16)e1;
      }
    }
    __builtin_amdgcn_wave_barrier();
    // Re-read P in A-fragment layout, then P @ V (V fragments shared by both tiles)
    v16bf aP0 = fragA(&pbuf[wv][0][nl][0], 0, half);
    v16bf aP1 = fragA(&pbuf[wv][1][nl][0], 0, half);
    #pragma unroll
    for (int t = 0; t < 4; t++) {
      v16bf bV = fragB(Vt + ((size_t)bh * 64 + (t * 16 + nl)) * Sn, j0, half);
      o[t]     = wmma_bf16(aP0, bV, o[t]);
      o[4 + t] = wmma_bf16(aP1, bV, o[4 + t]);
    }
  }

  const int bb = bh >> 4, h = bh & 15;
  #pragma unroll
  for (int qt = 0; qt < 2; qt++) {
    #pragma unroll
    for (int g = 0; g < 8; g++) {
      int row = i0 + qt * 16 + g + 8 * half;
      float rinv = 1.0f / lrow[qt * 8 + g];
      bf16* dr = Ab + ((size_t)(bb * Sn + row)) * HIDn + h * 64;
      dr[ 0 + nl] = (bf16)(o[qt * 4 + 0][g] * rinv);
      dr[16 + nl] = (bf16)(o[qt * 4 + 1][g] * rinv);
      dr[32 + nl] = (bf16)(o[qt * 4 + 2][g] * rinv);
      dr[48 + nl] = (bf16)(o[qt * 4 + 3][g] * rinv);
    }
  }
}

// ------------------------------------------- K4: output projection (fp32 out)
// grid(16, 16), block 256. Wave: 32 rows x 64 cols.
__global__ void k_out(const bf16* __restrict__ Ab, const bf16* __restrict__ WoT,
                      float* __restrict__ out) {
  const int lane = threadIdx.x & 31;
  const int wv   = threadIdx.x >> 5;
  const int half = lane >> 4;
  const int nl   = lane & 15;
  const int mt   = blockIdx.x * 8 + wv;    // 0..127
  const int n0   = blockIdx.y * 64;
  const int r0   = mt * 32;

  const bf16* arow0 = Ab + (size_t)(r0 + nl) * HIDn;
  const bf16* arow1 = Ab + (size_t)(r0 + 16 + nl) * HIDn;
  const bf16* brow[4];
  #pragma unroll
  for (int t = 0; t < 4; t++) brow[t] = WoT + (size_t)(n0 + t * 16 + nl) * HIDn;

  v8f acc[8] = {};
  for (int ko = 0; ko < HIDn; ko += 32) {
    v16bf a0 = fragA(arow0, ko, half);
    v16bf a1 = fragA(arow1, ko, half);
    #pragma unroll
    for (int t = 0; t < 4; t++) {
      v16bf b = fragB(brow[t], ko, half);
      acc[t]     = wmma_bf16(a0, b, acc[t]);
      acc[4 + t] = wmma_bf16(a1, b, acc[4 + t]);
    }
  }
  #pragma unroll
  for (int rt = 0; rt < 2; rt++) {
    #pragma unroll
    for (int g = 0; g < 8; g++) {
      float* dr = out + (size_t)(r0 + rt * 16 + g + 8 * half) * HIDn + n0;
      dr[ 0 + nl] = acc[rt * 4 + 0][g];
      dr[16 + nl] = acc[rt * 4 + 1][g];
      dr[32 + nl] = acc[rt * 4 + 2][g];
      dr[48 + nl] = acc[rt * 4 + 3][g];
    }
  }
}

extern "C" void kernel_launch(void* const* d_in, const int* in_sizes, int n_in,
                              void* d_out, int out_size, void* d_ws, size_t ws_size,
                              hipStream_t stream) {
  (void)in_sizes; (void)n_in; (void)out_size; (void)ws_size;
  const float* hidden = (const float*)d_in[0];
  const float* info   = (const float*)d_in[1];
  const float* Wq     = (const float*)d_in[2];
  const float* Wk     = (const float*)d_in[3];
  const float* Wv     = (const float*)d_in[4];
  const float* Wo     = (const float*)d_in[5];
  float* out = (float*)d_out;

  // Workspace carve-up (bf16 elements). Total ~51 MB.
  bf16* p = (bf16*)d_ws;
  bf16* Xbf = p;  p += (size_t)Mn * HIDn;       // 4096x1024
  bf16* WqT = p;  p += (size_t)HIDn * HIDn;
  bf16* WkT = p;  p += (size_t)HIDn * HIDn;
  bf16* WvT = p;  p += (size_t)HIDn * HIDn;
  bf16* WoT = p;  p += (size_t)HIDn * HIDn;
  bf16* Qb  = p;  p += (size_t)Bn * Hn * Sn * Dn;   // (B,H,S,D)
  bf16* Kb  = p;  p += (size_t)Bn * Hn * Sn * Dn;   // (B,H,S,D)
  bf16* Vt  = p;  p += (size_t)Bn * Hn * Dn * Sn;   // (B,H,D,S)
  bf16* Ab  = p;  p += (size_t)Mn * HIDn;           // (B*S, HID)

  k_cvt<<<4096, 256, 0, stream>>>(hidden, Xbf, Mn * HIDn);
  k_wT <<<4096, 256, 0, stream>>>(Wq, WqT);
  k_wT <<<4096, 256, 0, stream>>>(Wk, WkT);
  k_wT <<<4096, 256, 0, stream>>>(Wv, WvT);
  k_wT <<<4096, 256, 0, stream>>>(Wo, WoT);

  k_qkv <<<dim3(16, 16, 3), 256, 0, stream>>>(Xbf, WqT, WkT, WvT, info, Qb, Kb, Vt);
  k_attn<<<dim3(8, 32),     256, 0, stream>>>(Qb, Kb, Vt, Ab);
  k_out <<<dim3(16, 16),    256, 0, stream>>>(Ab, WoT, out);
}
